// MaskedCrossAttention_41884521070930
// MI455X (gfx1250) — matrix operand order repaired and checked by reading, not compiled, not run on hardware
//
#include <hip/hip_runtime.h>
#include <hip/hip_bf16.h>

typedef __attribute__((ext_vector_type(16))) _Float16 v16h;
typedef __attribute__((ext_vector_type(8)))  _Float16 v8h;
typedef __attribute__((ext_vector_type(4)))  _Float16 v4h;
typedef __attribute__((ext_vector_type(8)))  float    v8f;
typedef __attribute__((ext_vector_type(4)))  float    v4f;

#define B_     2
#define LQ_    2048
#define LK_    2048
#define DIM_   1024
#define HEADS_ 8
#define DH_    64
#define INNER_ 512

// ------------------------------------------------ gfx1250 async copy helpers
// global_load_async_to_lds_b128: VDST = per-lane LDS byte offset (as3 offset),
// VADDR = per-lane 64-bit global address.  Tracked by ASYNCcnt.
__device__ __forceinline__ void async_g2l_b128(unsigned lds_off, const void* gptr) {
  unsigned long long ga = (unsigned long long)(uintptr_t)gptr;
  asm volatile("global_load_async_to_lds_b128 %0, %1, off"
               :: "v"(lds_off), "v"(ga) : "memory");
}
__device__ __forceinline__ void wait_async0() {
  asm volatile("s_wait_asynccnt 0x0" ::: "memory");
}
__device__ __forceinline__ unsigned lds_off(const void* p) {
  return (unsigned)(uintptr_t)p;   // low 32 bits of generic LDS ptr = LDS offset
}

// ---------------------------------------------------------------- cast f32->f16
__global__ __launch_bounds__(256) void cast_f2h(const float* __restrict__ in,
                                                _Float16* __restrict__ out, int n4) {
  int i = blockIdx.x * blockDim.x + threadIdx.x;
  if (i < n4) {
    v4f f = ((const v4f*)in)[i];
    v4h h;
    h[0] = (_Float16)f[0]; h[1] = (_Float16)f[1];
    h[2] = (_Float16)f[2]; h[3] = (_Float16)f[3];
    ((v4h*)out)[i] = h;
  }
}

// ------------------------------------------------------- WMMA fragment loaders
// A fragment 16x32 (MxK) from row-major src: lanes 0-15 rows 0-15, K {0..7,16..23};
// lanes 16-31 same rows, K {8..15,24..31}.  Two contiguous 16B loads per lane.
__device__ __forceinline__ v16h load_A16x32(const _Float16* base, int ld, int l, int hi) {
  const _Float16* p = base + (size_t)l * ld + hi * 8;
  v8h lo  = *(const v8h*)p;
  v8h hi8 = *(const v8h*)(p + 16);
  v16h a;
#pragma unroll
  for (int i = 0; i < 8; ++i) { a[i] = lo[i]; a[8 + i] = hi8[i]; }
  return a;
}

// B fragment 32x16 (KxN) from row-major [N][K] view: lanes 0-15 col n=lane K {0..7,16..23},
// lanes 16-31 col n=lane-16 K {8..15,24..31}.  Two contiguous 16B loads per lane.
__device__ __forceinline__ v16h load_B32x16(const _Float16* baseT, int ld, int l, int hi) {
  const _Float16* p = baseT + (size_t)l * ld + hi * 8;
  v8h b0 = *(const v8h*)p;
  v8h b1 = *(const v8h*)(p + 16);
  v16h b;
#pragma unroll
  for (int i = 0; i < 8; ++i) { b[i] = b0[i]; b[8 + i] = b1[i]; }
  return b;
}

__device__ __forceinline__ v8f wmma16(v16h a, v16h b, v8f c) {
  return __builtin_amdgcn_wmma_f32_16x16x32_f16(false, a, false, b, (short)0, c, false, false);
}

// ------------------------------------------------------------ generic GEMM
// C[M,N] = A[M,K] @ Bt[N,K]^T * scale.  Block = 4 waves; wave tile 16(M) x 64(N).
// The 16xK A row-block (shared by all 4 waves) is async-staged into LDS once.
// mode 0: f16 row-major out; mode 1: f16 per-batch transposed out (V^T);
// mode 2: f32 row-major out.
__global__ __launch_bounds__(128) void gemm_wmma(const _Float16* __restrict__ A,
                                                 const _Float16* __restrict__ Bt,
                                                 void* __restrict__ out,
                                                 int M, int N, int K,
                                                 int mode, float scale, int LKdim) {
  __shared__ __align__(16) _Float16 atile[16 * 1024];    // up to K=1024 -> 32KB
  const int tid  = threadIdx.x;
  const int lane = tid & 31;
  const int wave = tid >> 5;
  const int l = lane & 15, hi = lane >> 4;
  const int m0 = blockIdx.x * 16;
  const int n0 = (blockIdx.y * 4 + wave) * 64;

  // ---- async stage A row-block [16][K] into LDS (16B chunks, all 128 lanes)
  const int cpr = K >> 3;                 // 16B chunks per row
  for (int c = tid; c < 2 * K; c += 128) {
    int row = c / cpr, part = c % cpr;
    async_g2l_b128(lds_off(&atile[row * K + part * 8]),
                   A + (size_t)(m0 + row) * K + part * 8);
  }
  wait_async0();
  __syncthreads();

  v8f acc[4] = {};
  for (int k = 0; k < K; k += 32) {
    v16h a = load_A16x32(atile + k, K, l, hi);           // ds_load_b128 x2
    if (k + 32 < K)
      __builtin_prefetch(Bt + (size_t)(n0 + l) * K + k + 32, 0, 3);
#pragma unroll
    for (int t = 0; t < 4; ++t) {
      v16h b = load_B32x16(Bt + (size_t)(n0 + t * 16) * K + k, K, l, hi);
      acc[t] = wmma16(a, b, acc[t]);
    }
  }

  if (mode == 0) {
    _Float16* o16 = (_Float16*)out;
#pragma unroll
    for (int t = 0; t < 4; ++t)
#pragma unroll
      for (int r = 0; r < 8; ++r) {
        int m = m0 + r + hi * 8;
        int n = n0 + t * 16 + l;
        o16[(size_t)m * N + n] = (_Float16)(acc[t][r] * scale);
      }
  } else if (mode == 1) {
    int bIdx = m0 / LKdim;
    int lk0  = (m0 % LKdim) + hi * 8;
#pragma unroll
    for (int t = 0; t < 4; ++t) {
      int n = n0 + t * 16 + l;
      _Float16* dst = (_Float16*)out + ((size_t)bIdx * N + n) * LKdim + lk0;
      v8h pk;
#pragma unroll
      for (int r = 0; r < 8; ++r) pk[r] = (_Float16)(acc[t][r] * scale);
      *(v8h*)dst = pk;              // 8 consecutive lk -> 16B store
    }
  } else {
    float* of = (float*)out;
#pragma unroll
    for (int t = 0; t < 4; ++t)
#pragma unroll
      for (int r = 0; r < 8; ++r) {
        int m = m0 + r + hi * 8;
        int n = n0 + t * 16 + l;
        of[(size_t)m * N + n] = acc[t][r] * scale;
      }
  }
}

// ------------------------------------------------------ flash attention (per b,h)
// Block = 4 waves sharing (b,h); each wave owns 16 query rows.  The 32-key
// K-tile (32x64) and V-tile (64x32) are shared -> double-buffered async LDS
// staging overlapped with the 8 WMMAs per chunk.  Online softmax in f32.
__global__ __launch_bounds__(128) void attn_wmma(const _Float16* __restrict__ q16,
                                                 const _Float16* __restrict__ k16,
                                                 const _Float16* __restrict__ vT,
                                                 const float* __restrict__ mask,
                                                 _Float16* __restrict__ att) {
  __shared__ __align__(16) _Float16 kbuf[2][32 * 64];   // [key][dh]
  __shared__ __align__(16) _Float16 vbuf[2][64 * 32];   // [dh][key]
  __shared__ __align__(16) _Float16 plds[4][16 * 32];   // per-wave P staging
  const int tid  = threadIdx.x;
  const int lane = tid & 31;
  const int wave = tid >> 5;
  const int l = lane & 15, hi = lane >> 4;
  const int q0 = (blockIdx.x * 4 + wave) * 16;
  const int h  = blockIdx.y;
  const int b  = blockIdx.z;

  const size_t qrow0 = (size_t)(b * LQ_ + q0);
  v16h qa0 = load_A16x32(q16 + qrow0 * INNER_ + h * DH_,      INNER_, l, hi);
  v16h qa1 = load_A16x32(q16 + qrow0 * INNER_ + h * DH_ + 32, INNER_, l, hi);

  v8f o0 = {}, o1 = {}, o2 = {}, o3 = {};
  float mrow[8], lrow[8];
#pragma unroll
  for (int r = 0; r < 8; ++r) { mrow[r] = -1e30f; lrow[r] = 0.f; }

  const float* mbase = mask + (size_t)b * LQ_ * LK_;
  _Float16* pw = plds[wave];

  // issue async copies of the 32-key chunk starting at key kk into buffer bb
  auto issue = [&](int kk, int bb) {
    // K tile: 256 x 16B chunks   (row = key, 64 dh)
#pragma unroll
    for (int j = 0; j < 2; ++j) {
      int c = tid + 128 * j;
      int key = c >> 3, part = c & 7;
      async_g2l_b128(lds_off(&kbuf[bb][key * 64 + part * 8]),
                     k16 + ((size_t)(b * LK_ + kk + key)) * INNER_ + h * DH_ + part * 8);
    }
    // V tile: 256 x 16B chunks   (row = dh, 32 keys)
#pragma unroll
    for (int j = 0; j < 2; ++j) {
      int c = tid + 128 * j;
      int d = c >> 2, part = c & 3;
      async_g2l_b128(lds_off(&vbuf[bb][d * 32 + part * 8]),
                     vT + ((size_t)b * INNER_ + h * DH_ + d) * LK_ + kk + part * 8);
    }
  };

  issue(0, 0);
  const int nIter = LK_ / 32;
  for (int it = 0; it < nIter; ++it) {
    const int bb = it & 1;
    const int kk = it * 32;
    wait_async0();          // this wave's copies of chunk `it` landed
    __syncthreads();        // all waves' copies landed; prev readers of bb^1 done
    if (it + 1 < nIter) issue(kk + 32, bb ^ 1);   // overlap next copy w/ compute

    const _Float16* kls = kbuf[bb];
    const _Float16* vls = vbuf[bb];

    // ---- S = Q @ K^T  (two 16-key subtiles, K-dim = 64) from LDS
    v8f c0 = {}, c1 = {};
    c0 = wmma16(qa0, load_B32x16(kls,                64, l, hi), c0);
    c0 = wmma16(qa1, load_B32x16(kls + 32,           64, l, hi), c0);
    c1 = wmma16(qa0, load_B32x16(kls + 16 * 64,      64, l, hi), c1);
    c1 = wmma16(qa1, load_B32x16(kls + 16 * 64 + 32, 64, l, hi), c1);

    // ---- additive mask (1 => -1e9)
#pragma unroll
    for (int r = 0; r < 8; ++r) {
      int qr = q0 + r + hi * 8;
      c0[r] += mbase[(size_t)qr * LK_ + kk + l]      * -1e9f;
      c1[r] += mbase[(size_t)qr * LK_ + kk + 16 + l] * -1e9f;
    }
    // ---- online softmax update (row reductions across 16-lane halves)
#pragma unroll
    for (int r = 0; r < 8; ++r) {
      float t = fmaxf(c0[r], c1[r]);
      t = fmaxf(t, __shfl_xor(t, 1, 32));
      t = fmaxf(t, __shfl_xor(t, 2, 32));
      t = fmaxf(t, __shfl_xor(t, 4, 32));
      t = fmaxf(t, __shfl_xor(t, 8, 32));
      float mn    = fmaxf(mrow[r], t);
      float alpha = __expf(mrow[r] - mn);
      mrow[r] = mn;
      float e0 = __expf(c0[r] - mn);
      float e1 = __expf(c1[r] - mn);
      c0[r] = e0; c1[r] = e1;
      float s = e0 + e1;
      s += __shfl_xor(s, 1, 32);
      s += __shfl_xor(s, 2, 32);
      s += __shfl_xor(s, 4, 32);
      s += __shfl_xor(s, 8, 32);
      lrow[r] = lrow[r] * alpha + s;
      o0[r] *= alpha; o1[r] *= alpha; o2[r] *= alpha; o3[r] *= alpha;
    }
    // ---- reshape P (D-layout) -> A-fragment via per-wave LDS tile
#pragma unroll
    for (int r = 0; r < 8; ++r) {
      int m = r + hi * 8;
      pw[m * 32 + l]      = (_Float16)c0[r];
      pw[m * 32 + 16 + l] = (_Float16)c1[r];
    }
    asm volatile("s_wait_dscnt 0" ::: "memory");   // intra-wave LDS RAW
    v16h pa = load_A16x32(pw, 32, l, hi);
    // ---- O += P @ V  from LDS (K-dim = 32 keys)
    o0 = wmma16(pa, load_B32x16(vls +  0 * 32, 32, l, hi), o0);
    o1 = wmma16(pa, load_B32x16(vls + 16 * 32, 32, l, hi), o1);
    o2 = wmma16(pa, load_B32x16(vls + 32 * 32, 32, l, hi), o2);
    o3 = wmma16(pa, load_B32x16(vls + 48 * 32, 32, l, hi), o3);
  }
  // ---- normalize and store row-major f16 [B*LQ, INNER]
#pragma unroll
  for (int r = 0; r < 8; ++r) {
    float inv = 1.0f / lrow[r];
    int qr = q0 + r + hi * 8;
    size_t off = ((size_t)(b * LQ_ + qr)) * INNER_ + h * DH_ + l;
    att[off +  0] = (_Float16)(o0[r] * inv);
    att[off + 16] = (_Float16)(o1[r] * inv);
    att[off + 32] = (_Float16)(o2[r] * inv);
    att[off + 48] = (_Float16)(o3[r] * inv);
  }
}

// ---------------------------------------------------------------- launcher
extern "C" void kernel_launch(void* const* d_in, const int* in_sizes, int n_in,
                              void* d_out, int out_size, void* d_ws, size_t ws_size,
                              hipStream_t stream) {
  const float* x    = (const float*)d_in[0];
  const float* y    = (const float*)d_in[1];
  const float* mask = (const float*)d_in[2];
  const float* Wq   = (const float*)d_in[3];
  const float* Wk   = (const float*)d_in[4];
  const float* Wv   = (const float*)d_in[5];
  const float* Wo   = (const float*)d_in[6];

  char* ws = (char*)d_ws;
  size_t off = 0;
  auto alloc = [&](size_t bytes) -> void* {
    void* p = ws + off;
    off += (bytes + 255) & ~(size_t)255;
    return p;
  };
  _Float16* x16  = (_Float16*)alloc((size_t)B_ * LQ_ * DIM_ * 2);
  _Float16* y16  = (_Float16*)alloc((size_t)B_ * LK_ * DIM_ * 2);
  _Float16* wq16 = (_Float16*)alloc((size_t)INNER_ * DIM_ * 2);
  _Float16* wk16 = (_Float16*)alloc((size_t)INNER_ * DIM_ * 2);
  _Float16* wv16 = (_Float16*)alloc((size_t)INNER_ * DIM_ * 2);
  _Float16* wo16 = (_Float16*)alloc((size_t)DIM_ * INNER_ * 2);
  _Float16* q16  = (_Float16*)alloc((size_t)B_ * LQ_ * INNER_ * 2);
  _Float16* k16  = (_Float16*)alloc((size_t)B_ * LK_ * INNER_ * 2);
  _Float16* vT16 = (_Float16*)alloc((size_t)B_ * INNER_ * LK_ * 2);
  _Float16* a16  = (_Float16*)alloc((size_t)B_ * LQ_ * INNER_ * 2);
  if (off > ws_size) return;

  auto cast = [&](const float* in, _Float16* out, size_t n) {
    int n4 = (int)(n / 4);
    cast_f2h<<<(n4 + 255) / 256, 256, 0, stream>>>(in, out, n4);
  };
  cast(x,  x16,  (size_t)B_ * LQ_ * DIM_);
  cast(y,  y16,  (size_t)B_ * LK_ * DIM_);
  cast(Wq, wq16, (size_t)INNER_ * DIM_);
  cast(Wk, wk16, (size_t)INNER_ * DIM_);
  cast(Wv, wv16, (size_t)INNER_ * DIM_);
  cast(Wo, wo16, (size_t)DIM_ * INNER_);

  const int M = B_ * LQ_;                 // 4096
  const float qscale = 0.125f;            // DH^-0.5

  // Q = x @ Wq^T * scale       [4096,512]  f16 row-major
  gemm_wmma<<<dim3(M / 16, INNER_ / 256), 128, 0, stream>>>(
      x16, wq16, q16, M, INNER_, DIM_, 0, qscale, LK_);
  // K = y @ Wk^T               [4096,512]  f16 row-major
  gemm_wmma<<<dim3(M / 16, INNER_ / 256), 128, 0, stream>>>(
      y16, wk16, k16, M, INNER_, DIM_, 0, 1.0f, LK_);
  // V^T = (y @ Wv^T)^T per batch  [B*INNER, LK]  f16
  gemm_wmma<<<dim3(M / 16, INNER_ / 256), 128, 0, stream>>>(
      y16, wv16, vT16, M, INNER_, DIM_, 1, 1.0f, LK_);
  // fused masked softmax attention -> a16 [4096,512] f16
  attn_wmma<<<dim3(LQ_ / 64, HEADS_, B_), 128, 0, stream>>>(
      q16, k16, vT16, mask, a16);
  // out = a @ Wo^T             [4096,1024] f32
  gemm_wmma<<<dim3(M / 16, DIM_ / 256), 128, 0, stream>>>(
      a16, wo16, (float*)d_out, M, DIM_, INNER_, 2, 1.0f, LK_);
}
